// SiglipAttention_16252156248250
// MI455X (gfx1250) — matrix operand order repaired and checked by reading, not compiled, not run on hardware
//
#include <hip/hip_runtime.h>
#include <hip/hip_bf16.h>

typedef __attribute__((ext_vector_type(16))) _Float16 v16h;
typedef __attribute__((ext_vector_type(8)))  _Float16 v8h;
typedef __attribute__((ext_vector_type(8)))  float    v8f;
typedef __attribute__((ext_vector_type(4)))  float    v4f;

#define D_MODEL 1152
#define SEQ     729
#define BATCH   16
#define NHEADS  16
#define HDIM    72
#define SP      736            // SEQ padded to 23*32
#define HDP     80             // head dim padded to 5*16
#define QTILES  46             // ceil(729/16)
#define NROWS   (BATCH*SEQ)    // 11664

static __device__ __forceinline__ v8f wmma16(v16h a, v16h b, v8f c) {
  return __builtin_amdgcn_wmma_f32_16x16x32_f16(false, a, false, b, (short)0, c,
                                                false, false);
}
static __device__ __forceinline__ v4f ld4f(const float* p) { return *(const v4f*)p; }
static __device__ __forceinline__ v8h ld8h(const _Float16* p) { return *(const v8h*)p; }
static __device__ __forceinline__ v8h cvt8(v4f a, v4f b) {
  v8h r;
  r[0]=(_Float16)a[0]; r[1]=(_Float16)a[1]; r[2]=(_Float16)a[2]; r[3]=(_Float16)a[3];
  r[4]=(_Float16)b[0]; r[5]=(_Float16)b[1]; r[6]=(_Float16)b[2]; r[7]=(_Float16)b[3];
  return r;
}
static __device__ __forceinline__ v16h cat16(v8h lo, v8h hi) {
  v16h r;
#pragma unroll
  for (int i = 0; i < 8; ++i) { r[i] = lo[i]; r[i+8] = hi[i]; }
  return r;
}

// ---------------------------------------------------------------------------
// Stage 1: q,k,v = x @ W^T + b   (A tile shared by 3 WMMA chains per K-chunk)
// q,k -> f16 [NROWS, D_MODEL]; v -> f16 transposed [B*H, HDP, SP] (zero-padded)
// ---------------------------------------------------------------------------
__global__ __launch_bounds__(128)
void qkv_kernel(const float* __restrict__ x,
                const float* __restrict__ qw, const float* __restrict__ qb,
                const float* __restrict__ kw, const float* __restrict__ kb,
                const float* __restrict__ vw, const float* __restrict__ vb,
                _Float16* __restrict__ qh, _Float16* __restrict__ kh,
                _Float16* __restrict__ vT)
{
  const int wid   = blockIdx.x * 4 + (threadIdx.x >> 5);
  const int lane  = threadIdx.x & 31;
  const int lhalf = lane >> 4, lmod = lane & 15;
  const int mtile = wid / (D_MODEL/16);
  const int ntile = wid % (D_MODEL/16);
  const int n     = ntile*16 + lmod;
  const int arow  = mtile*16 + lmod;

  const float* xa  = x  + (size_t)arow * D_MODEL + lhalf*16;
  const float* qwp = qw + (size_t)n    * D_MODEL + lhalf*16;
  const float* kwp = kw + (size_t)n    * D_MODEL + lhalf*16;
  const float* vwp = vw + (size_t)n    * D_MODEL + lhalf*16;

  v8f aq = {}, ak = {}, av = {};
  for (int kc = 0; kc < D_MODEL/32; ++kc) {
    const float* xp = xa + kc*32;
    __builtin_prefetch(xp + 64, 0, 3);   // global_prefetch_b8, next-next chunk
    v16h A = cat16(cvt8(ld4f(xp),      ld4f(xp + 4)),
                   cvt8(ld4f(xp + 8), ld4f(xp + 12)));
    {
      const float* wp = qwp + kc*32;
      v16h Bm = cat16(cvt8(ld4f(wp), ld4f(wp+4)), cvt8(ld4f(wp+8), ld4f(wp+12)));
      aq = wmma16(A, Bm, aq);
    }
    {
      const float* wp = kwp + kc*32;
      v16h Bm = cat16(cvt8(ld4f(wp), ld4f(wp+4)), cvt8(ld4f(wp+8), ld4f(wp+12)));
      ak = wmma16(A, Bm, ak);
    }
    {
      const float* wp = vwp + kc*32;
      v16h Bm = cat16(cvt8(ld4f(wp), ld4f(wp+4)), cvt8(ld4f(wp+8), ld4f(wp+12)));
      av = wmma16(A, Bm, av);
    }
  }

  const float scale = 0.11785113019775793f;  // 72^-0.5, folded into q
  const float bq = qb[n], bk = kb[n], bv = vb[n];
  const int h = n / HDIM, d = n - h * HDIM;
#pragma unroll
  for (int j = 0; j < 8; ++j) {
    const int row = mtile*16 + j + 8*lhalf;           // M = j + 8*(lane/16)
    qh[(size_t)row * D_MODEL + n] = (_Float16)((aq[j] + bq) * scale);
    kh[(size_t)row * D_MODEL + n] = (_Float16)(ak[j] + bk);
    const int b = row / SEQ, s = row - b * SEQ;
    vT[((size_t)(b*NHEADS + h) * HDP + d) * SP + s] = (_Float16)(av[j] + bv);
  }
}

// ---------------------------------------------------------------------------
// Stage 2 (fused): scores = q k^T -> LDS -> softmax -> attn (fp32 out) and
// ctx = attn @ v from the LDS probability strip (no HBM re-read).
// TWO waves cooperate on one 16x736 f32 strip (47 KB LDS) => ~3 waves/SIMD.
// Phase A: waves split key tiles; Phase B: waves split rows; Phase C: waves
// compute overlapping d-tile triples (middle tile redundant) so no WMMA ever
// sits under divergent control flow (EXEC must be all-ones for WMMA).
// ---------------------------------------------------------------------------
__global__ __launch_bounds__(64)
void attn_ctx_kernel(const _Float16* __restrict__ qh,
                     const _Float16* __restrict__ kh,
                     const _Float16* __restrict__ vT,
                     float* __restrict__ attn,
                     _Float16* __restrict__ ctx)
{
  __shared__ float sc[16 * SP];
  const int w     = threadIdx.x >> 5;       // wave id 0/1 (uniform per wave)
  const int lane  = threadIdx.x & 31;
  const int lhalf = lane >> 4, lmod = lane & 15;
  const int qtile = blockIdx.x % QTILES;
  const int bh    = blockIdx.x / QTILES;
  const int b     = bh >> 4, h = bh & 15;

  // ---- phase A: scores into LDS (waves interleave key tiles) -------------
  int sq = qtile*16 + lmod; if (sq > SEQ-1) sq = SEQ-1;
  const _Float16* qp = qh + (size_t)(b*SEQ + sq) * D_MODEL + h*HDIM;

  const v16h A0 = cat16(ld8h(qp + lhalf*16),      ld8h(qp + lhalf*16 + 8));
  const v16h A1 = cat16(ld8h(qp + 32 + lhalf*16), ld8h(qp + 32 + lhalf*16 + 8));
  v16h A2;
  {
    v8h lo = {}; v8h hi = {};
    if (lhalf == 0) lo = ld8h(qp + 64);   // d = 64..71; rest of K padded 0
    A2 = cat16(lo, hi);
  }

  for (int nt = w; nt < QTILES; nt += 2) {
    int sk = nt*16 + lmod; if (sk > SEQ-1) sk = SEQ-1;
    const _Float16* kp = kh + (size_t)(b*SEQ + sk) * D_MODEL + h*HDIM;
    v16h B0 = cat16(ld8h(kp + lhalf*16),      ld8h(kp + lhalf*16 + 8));
    v16h B1 = cat16(ld8h(kp + 32 + lhalf*16), ld8h(kp + 32 + lhalf*16 + 8));
    v16h B2;
    {
      v8h lo = {}; v8h hi = {};
      if (lhalf == 0) lo = ld8h(kp + 64);
      B2 = cat16(lo, hi);
    }
    v8f acc = {};
    acc = wmma16(A0, B0, acc);
    acc = wmma16(A1, B1, acc);
    acc = wmma16(A2, B2, acc);
#pragma unroll
    for (int j = 0; j < 8; ++j)
      sc[(j + 8*lhalf) * SP + nt*16 + lmod] = acc[j];
  }
  __syncthreads();

  // ---- phase B: softmax (wave w owns rows 8w..8w+7); probs kept in LDS ---
  for (int r = w*8; r < w*8 + 8; ++r) {
    float* row = sc + r * SP;
    float m = -3.4e38f;
    for (int c = lane; c < SEQ; c += 32) m = fmaxf(m, row[c]);
#pragma unroll
    for (int off = 16; off >= 1; off >>= 1) m = fmaxf(m, __shfl_xor(m, off, 32));
    float ssum = 0.f;
    for (int c = lane; c < SEQ; c += 32) {
      float e = __expf(row[c] - m);
      row[c] = e;
      ssum += e;
    }
#pragma unroll
    for (int off = 16; off >= 1; off >>= 1) ssum += __shfl_xor(ssum, off, 32);
    const float inv = 1.0f / ssum;
    const int srow = qtile*16 + r;
    if (srow < SEQ) {
      float* op = attn + ((size_t)(bh*SEQ + srow)) * SEQ;
      for (int c = lane; c < SEQ; c += 32) {
        const float p = row[c] * inv;
        row[c] = p;               // keep normalized prob for the attn@v phase
        op[c]  = p;               // mandatory fp32 attn output
      }
    } else {
      for (int c = lane; c < SEQ; c += 32) row[c] *= inv;
    }
    if (lane < SP - SEQ) row[SEQ + lane] = 0.f;   // zero K-padding columns
  }
  __syncthreads();

  // ---- phase C: ctx = attn @ v from LDS ----------------------------------
  // wave0 -> d-tiles {0,1,2}, wave1 -> d-tiles {2,3,4}; tile 2 redundant,
  // stored only by wave0. Unconditional WMMA in both waves.
  const int dtbase = w * 2;
  v8f acc0 = {}, acc1 = {}, acc2 = {};
  const _Float16* vbase =
      vT + ((size_t)bh * HDP + dtbase*16 + lmod) * SP + lhalf*16;
  for (int kc = 0; kc < SP/32; ++kc) {
    const int k0 = kc * 32;
    const float* ar = sc + lmod * SP + k0 + lhalf*16;          // 32B aligned
    const v16h A = cat16(cvt8(ld4f(ar),      ld4f(ar + 4)),
                         cvt8(ld4f(ar + 8), ld4f(ar + 12)));
    const _Float16* bp0 = vbase + k0;
    const _Float16* bp1 = bp0 + (size_t)16 * SP;
    const _Float16* bp2 = bp1 + (size_t)16 * SP;
    acc0 = wmma16(A, cat16(ld8h(bp0), ld8h(bp0 + 8)), acc0);
    acc1 = wmma16(A, cat16(ld8h(bp1), ld8h(bp1 + 8)), acc1);
    acc2 = wmma16(A, cat16(ld8h(bp2), ld8h(bp2 + 8)), acc2);
  }

#pragma unroll
  for (int t = 0; t < 3; ++t) {
    const v8f acc = (t == 0) ? acc0 : (t == 1) ? acc1 : acc2;
    const int dt = dtbase + t;
    const int d  = dt*16 + lmod;
    const bool own = (w == 0) ? (t < 3) : (t > 0);   // tile 2 stored by wave0
    if (own && d < HDIM) {
#pragma unroll
      for (int j = 0; j < 8; ++j) {
        const int s = qtile*16 + j + 8*lhalf;
        if (s < SEQ)
          ctx[(size_t)(b*SEQ + s) * D_MODEL + h*HDIM + d] = (_Float16)acc[j];
      }
    }
  }
}

// ---------------------------------------------------------------------------
// Stage 3: out = ctx @ o_w^T + o_b  (fp32 output)
// ---------------------------------------------------------------------------
__global__ __launch_bounds__(128)
void out_kernel(const _Float16* __restrict__ ctx, const float* __restrict__ ow,
                const float* __restrict__ ob, float* __restrict__ out)
{
  const int wid   = blockIdx.x * 4 + (threadIdx.x >> 5);
  const int lane  = threadIdx.x & 31;
  const int lhalf = lane >> 4, lmod = lane & 15;
  const int mtile = wid / (D_MODEL/16);
  const int ntile = wid % (D_MODEL/16);
  const int n     = ntile*16 + lmod;

  const _Float16* ap = ctx + (size_t)(mtile*16 + lmod) * D_MODEL + lhalf*16;
  const float*    wp = ow  + (size_t)n * D_MODEL + lhalf*16;

  v8f acc = {};
  for (int kc = 0; kc < D_MODEL/32; ++kc) {
    const _Float16* a = ap + kc*32;
    const v16h A = cat16(ld8h(a), ld8h(a + 8));
    const float* w = wp + kc*32;
    const v16h B = cat16(cvt8(ld4f(w), ld4f(w+4)), cvt8(ld4f(w+8), ld4f(w+12)));
    acc = wmma16(A, B, acc);
  }
  const float bn = ob[n];
#pragma unroll
  for (int j = 0; j < 8; ++j) {
    const int row = mtile*16 + j + 8*lhalf;
    out[(size_t)row * D_MODEL + n] = acc[j] + bn;
  }
}

// ---------------------------------------------------------------------------
extern "C" void kernel_launch(void* const* d_in, const int* in_sizes, int n_in,
                              void* d_out, int out_size, void* d_ws, size_t ws_size,
                              hipStream_t stream)
{
  const float* x  = (const float*)d_in[0];
  const float* qw = (const float*)d_in[1];
  const float* qb = (const float*)d_in[2];
  const float* kw = (const float*)d_in[3];
  const float* kb = (const float*)d_in[4];
  const float* vw = (const float*)d_in[5];
  const float* vb = (const float*)d_in[6];
  const float* ow = (const float*)d_in[7];
  const float* ob = (const float*)d_in[8];

  float* out  = (float*)d_out;                          // [16,729,1152]
  float* attn = out + (size_t)BATCH * SEQ * D_MODEL;    // [16,16,729,729]

  char* ws = (char*)d_ws;
  const size_t QH_BYTES = (size_t)NROWS * D_MODEL * sizeof(_Float16);          // 26.9 MB
  const size_t VT_BYTES = (size_t)BATCH*NHEADS * HDP * SP * sizeof(_Float16);  // 30.1 MB
  _Float16* qh  = (_Float16*)(ws);
  _Float16* kh  = (_Float16*)(ws + QH_BYTES);
  _Float16* vT  = (_Float16*)(ws + 2*QH_BYTES);
  _Float16* ctx = (_Float16*)(ws + 2*QH_BYTES + VT_BYTES);  // ~111 MB total

  // zero padded regions of vT (seq cols 729..735, head rows 72..79)
  hipMemsetAsync(vT, 0, VT_BYTES, stream);

  {   // stage 1: 729*72 = 52488 tiles, 4 waves / block
    const int waves = (NROWS/16) * (D_MODEL/16);
    qkv_kernel<<<waves/4, 128, 0, stream>>>(x, qw, qb, kw, kb, vw, vb, qh, kh, vT);
  }
  {   // stage 2 (fused scores+softmax+attn@v): two waves per (b,h,qtile)
    const int blocks = BATCH * NHEADS * QTILES;           // 11776
    attn_ctx_kernel<<<blocks, 64, 0, stream>>>(qh, kh, vT, attn, ctx);
  }
  {   // stage 3: output projection
    const int waves = (NROWS/16) * (D_MODEL/16);
    out_kernel<<<waves/4, 128, 0, stream>>>(ctx, ow, ob, out);
  }
}
